// L0IndexedAttention_25469156065587
// MI455X (gfx1250) — compile-verified, matrix operand order checked by reading, not run on hardware
//
#include <hip/hip_runtime.h>
#include <hip/hip_bf16.h>

// ---------------------------------------------------------------------------
// L0IndexedAttention for MI455X (gfx1250, wave32, WMMA f32_16x16x32_f16)
// Memory-bound (~54 GFLOP vs ~270MB HBM traffic) -> f16 WMMA for all GEMMs,
// per-group bias scatter into LDS, TDM (tensor_load_to_lds) staging of K and
// V^T tiles in the attention kernel. Workspace use: ~180 MB.
// ---------------------------------------------------------------------------

typedef _Float16 half_t;
typedef __attribute__((ext_vector_type(16))) _Float16 v16h;
typedef __attribute__((ext_vector_type(8)))  float    v8f;
typedef __attribute__((ext_vector_type(4)))  unsigned int u32x4;
typedef __attribute__((ext_vector_type(8)))  int          i32x8;
typedef __attribute__((ext_vector_type(4)))  int          i32x4;

static constexpr int Gc   = 256;
static constexpr int Dc   = 512;
static constexpr int Hc   = 8;
static constexpr int HDc  = 64;
static constexpr int RBFc = 64;
static constexpr int DEGc = 32;
static constexpr int MAXCc = 128;
static constexpr int KQVc  = 3 * Dc;   // 1536

#define WMMA16(a, b, c) \
  __builtin_amdgcn_wmma_f32_16x16x32_f16(false, (a), false, (b), (short)0, (c), false, false)

#if __has_builtin(__builtin_amdgcn_tensor_load_to_lds) && \
    __has_builtin(__builtin_amdgcn_s_wait_tensorcnt)
#define HAVE_TDM 1
#endif

__device__ inline v8f v8f_zero() {
  v8f z = {0.f, 0.f, 0.f, 0.f, 0.f, 0.f, 0.f, 0.f};
  return z;
}

// A/B fragment, row-major source: element (row0 + lane&15, k0 + kmap(i)).
// kmap: lanes 0-15 -> K {0..7,16..23}; lanes 16-31 -> K {8..15,24..31}.
__device__ inline v16h ldfrag_row_f16(const half_t* p0, int ld, int lane) {
  const half_t* p = p0 + (size_t)(lane & 15) * ld + ((lane >> 4) << 3);
  v16h f;
#pragma unroll
  for (int i = 0; i < 8; ++i) { f[i] = p[i]; f[i + 8] = p[i + 16]; }
  return f;
}

#ifdef HAVE_TDM
// One-shot 1-D TDM DMA: contiguous f16 region global -> LDS.
// D# group0: count=1 | lds_addr | global_addr | type=2 ("image").
// D# group1: data_size=1(2B); tensor_dim0 = tile_dim0 = nelem; dim1=1.
__device__ inline void tdm_load_contig_f16(const half_t* gsrc, void* ldst,
                                           unsigned nelem) {
  unsigned long long ga = (unsigned long long)gsrc;
  unsigned la = (unsigned)(unsigned long long)ldst;  // generic addr low 32b = LDS offset
  u32x4 g0 = {1u, la, (unsigned)ga, (unsigned)(ga >> 32) | (2u << 30)};
  i32x8 g1 = {(int)(1u << 16),                              // data_size = 2 bytes
              (int)((nelem & 0xFFFFu) << 16),               // tensor_dim0[15:0]
              (int)(((nelem >> 16) & 0xFFFFu) | (1u << 16)),// tensor_dim0 hi | tensor_dim1=1
              (int)((nelem & 0xFFFFu) << 16),               // tile_dim0
              0,                                            // tile_dim1/2 unused
              (int)nelem,                                   // tensor_dim0_stride lo
              0, 0};
  i32x4 z4 = {0, 0, 0, 0};
#if __clang_major__ >= 23
  i32x8 z8 = {0, 0, 0, 0, 0, 0, 0, 0};
  __builtin_amdgcn_tensor_load_to_lds(g0, g1, z4, z4, z8, 0);
#else
  __builtin_amdgcn_tensor_load_to_lds(g0, g1, z4, z4, 0);
#endif
}
#endif

// ---------------------------------------------------------------------------
// counts[g] = popcount(mask row), offsets = exclusive scan (G=256, 1 block)
// ---------------------------------------------------------------------------
__global__ __launch_bounds__(256) void counts_offsets_kernel(
    const unsigned char* __restrict__ mask, int* __restrict__ counts,
    int* __restrict__ offsets) {
  __shared__ int cs[Gc];
  int g = threadIdx.x;
  int c = 0;
#pragma unroll 4
  for (int i = 0; i < MAXCc; ++i) c += mask[g * MAXCc + i] ? 1 : 0;
  cs[g] = c;
  counts[g] = c;
  __syncthreads();
  if (threadIdx.x == 0) {
    int acc = 0;
    for (int i = 0; i < Gc; ++i) { offsets[i] = acc; acc += cs[i]; }
  }
}

__global__ __launch_bounds__(256) void f32_to_f16_kernel(
    const float* __restrict__ in, half_t* __restrict__ out, long n) {
  long i = (long)blockIdx.x * 256 + threadIdx.x;
  if (i < n) out[i] = (half_t)in[i];
}

// ---------------------------------------------------------------------------
// LayerNorm over D=512, one row per 256-thread block, write f16
// ---------------------------------------------------------------------------
__global__ __launch_bounds__(256) void ln_features_kernel(
    const float* __restrict__ f, const float* __restrict__ g,
    const float* __restrict__ b, half_t* __restrict__ out) {
  const int row = blockIdx.x, tid = threadIdx.x;
  const float* x = f + (size_t)row * Dc;
  float x0 = x[tid], x1 = x[tid + 256];
  float s = x0 + x1, sq = x0 * x0 + x1 * x1;
  __shared__ float red[16];
#pragma unroll
  for (int o = 16; o > 0; o >>= 1) { s += __shfl_xor(s, o, 32); sq += __shfl_xor(sq, o, 32); }
  int wave = tid >> 5, lane = tid & 31;
  if (lane == 0) { red[wave] = s; red[wave + 8] = sq; }
  __syncthreads();
  if (wave == 0) {
    float ss = (lane < 8) ? red[lane] : 0.f;
    float qq = (lane < 8) ? red[lane + 8] : 0.f;
#pragma unroll
    for (int o = 4; o > 0; o >>= 1) { ss += __shfl_xor(ss, o, 32); qq += __shfl_xor(qq, o, 32); }
    if (lane == 0) { red[0] = ss; red[1] = qq; }
  }
  __syncthreads();
  float mu = red[0] * (1.f / Dc);
  float var = red[1] * (1.f / Dc) - mu * mu;
  float rs = rsqrtf(var + 1e-5f);
  out[(size_t)row * Dc + tid]       = (half_t)((x0 - mu) * rs * g[tid] + b[tid]);
  out[(size_t)row * Dc + tid + 256] = (half_t)((x1 - mu) * rs * g[tid + 256] + b[tid + 256]);
}

// ---------------------------------------------------------------------------
// kqv = xln @ w_kqv^T, scattered into padded k/q [G,H,MAXC,HD] and
// TRANSPOSED v [G,H,HD,MAXC] (all f16).
// ---------------------------------------------------------------------------
__global__ __launch_bounds__(256) void kqv_gemm_scatter_kernel(
    const half_t* __restrict__ xln, const half_t* __restrict__ wkqv,
    const int* __restrict__ batch_map, const int* __restrict__ pos_in_group,
    half_t* __restrict__ kpad, half_t* __restrict__ qpad, half_t* __restrict__ vpadT) {
  const int nstrips = KQVc / 64;                 // 24
  const int strip = blockIdx.x % nstrips;
  const int mblk  = blockIdx.x / nstrips;
  const int wave = threadIdx.x >> 5, lane = threadIdx.x & 31;
  const int m0 = mblk * 128 + wave * 16;
  const int n0 = strip * 64;

  v8f acc[4] = {v8f_zero(), v8f_zero(), v8f_zero(), v8f_zero()};
  for (int k = 0; k < Dc; k += 32) {
    __builtin_prefetch(xln + (size_t)m0 * Dc + k + 64, 0, 1);
    v16h a = ldfrag_row_f16(xln + (size_t)m0 * Dc + k, Dc, lane);
#pragma unroll
    for (int j = 0; j < 4; ++j) {
      v16h b = ldfrag_row_f16(wkqv + (size_t)(n0 + j * 16) * Dc + k, Dc, lane);
      acc[j] = WMMA16(a, b, acc[j]);
    }
  }
  const int col16 = lane & 15, hi = lane >> 4;
#pragma unroll
  for (int i = 0; i < 8; ++i) {
    const int m = m0 + i + 8 * hi;               // ragged row index n
    const int g = batch_map[m], p = pos_in_group[m];
#pragma unroll
    for (int j = 0; j < 4; ++j) {
      const int jj = n0 + j * 16 + col16;        // column in [0,1536)
      const int which = jj >> 9;                 // 0:k 1:q 2:v (torch chunk order)
      const int h = (jj & 511) >> 6, dc = jj & 63;
      const half_t val = (half_t)acc[j][i];
      if (which == 2) {
        vpadT[(((size_t)g * Hc + h) * HDc + dc) * MAXCc + p] = val;
      } else {
        half_t* dst = (which == 0) ? kpad : qpad;
        dst[(((size_t)g * Hc + h) * MAXCc + p) * HDc + dc] = val;
      }
    }
  }
}

// ---------------------------------------------------------------------------
// eb[e,h] = LN(edge_radial[e]) @ w_bias[h] + b_bias[h].  One wave per edge.
// ---------------------------------------------------------------------------
__global__ __launch_bounds__(256) void edge_bias_kernel(
    const float* __restrict__ er, const float* __restrict__ lg,
    const float* __restrict__ lb, const float* __restrict__ wb,
    const float* __restrict__ bb, float* __restrict__ eb, long E) {
  const int wave = threadIdx.x >> 5, lane = threadIdx.x & 31;
  const long e = (long)blockIdx.x * 8 + wave;
  if (e >= E) return;
  const float* x = er + e * RBFc;
  float x0 = x[lane], x1 = x[lane + 32];
  float s = x0 + x1, sq = x0 * x0 + x1 * x1;
#pragma unroll
  for (int o = 16; o > 0; o >>= 1) { s += __shfl_xor(s, o, 32); sq += __shfl_xor(sq, o, 32); }
  float mu = s * (1.f / RBFc), var = sq * (1.f / RBFc) - mu * mu;
  float rs = rsqrtf(var + 1e-5f);
  float y0 = (x0 - mu) * rs * lg[lane] + lb[lane];
  float y1 = (x1 - mu) * rs * lg[lane + 32] + lb[lane + 32];
#pragma unroll
  for (int h = 0; h < Hc; ++h) {
    float p = y0 * wb[h * RBFc + lane] + y1 * wb[h * RBFc + lane + 32];
#pragma unroll
    for (int o = 16; o > 0; o >>= 1) p += __shfl_xor(p, o, 32);
    if (lane == 0) eb[e * Hc + h] = p + bb[h];
  }
}

// ---------------------------------------------------------------------------
// Per-(g,h) attention block.
//   0. TDM-stage K tile (128x64) and V^T tile (64x128) into LDS
//   1. zero S; 2. scatter edge bias (set semantics); 3. S += QK^T/8 (WMMA);
//   4. masked softmax -> P16 (f16 LDS); 5. O = P16 V (WMMA) -> opad f16.
// LDS: 64 + 32 + 16 + 16 = 128 KB  (2 blocks / 320KB WGP)
// ---------------------------------------------------------------------------
__global__ __launch_bounds__(256) void attention_kernel(
    const half_t* __restrict__ qpad, const half_t* __restrict__ kpad,
    const half_t* __restrict__ vpadT, const float* __restrict__ eb,
    const int* __restrict__ lsrc, const int* __restrict__ ltgt,
    const int* __restrict__ counts, const int* __restrict__ offsets,
    half_t* __restrict__ opad) {
  __shared__ float  S[MAXCc][MAXCc];             // 64 KB scores
  __shared__ half_t P16[MAXCc][MAXCc];           // 32 KB probabilities
  __shared__ half_t Ksh[MAXCc * HDc];            // 16 KB K tile
  __shared__ half_t Vtsh[HDc * MAXCc];           // 16 KB V^T tile
  const int g = blockIdx.x / Hc, h = blockIdx.x % Hc;
  const int tid = threadIdx.x, wave = tid >> 5, lane = tid & 31;
  const int cnt = counts[g];
  const long e0 = (long)DEGc * offsets[g];
  const long e1 = e0 + (long)DEGc * cnt;

  const half_t* Qb  = qpad  + ((size_t)g * Hc + h) * MAXCc * HDc;
  const half_t* Kb  = kpad  + ((size_t)g * Hc + h) * MAXCc * HDc;
  const half_t* Vtb = vpadT + ((size_t)g * Hc + h) * HDc * MAXCc;

#ifdef HAVE_TDM
  if (tid < 32) {                                // one wave issues both DMAs
    tdm_load_contig_f16(Kb, Ksh, MAXCc * HDc);
    tdm_load_contig_f16(Vtb, Vtsh, HDc * MAXCc);
  }
#else
  {
    const uint4* ks = (const uint4*)Kb;  uint4* kd = (uint4*)Ksh;
    const uint4* vs = (const uint4*)Vtb; uint4* vd = (uint4*)Vtsh;
    for (int i = tid; i < MAXCc * HDc / 8; i += 256) { kd[i] = ks[i]; vd[i] = vs[i]; }
  }
#endif

  for (int i = tid; i < MAXCc * MAXCc; i += 256) (&S[0][0])[i] = 0.f;
  __syncthreads();

  for (long e = e0 + tid; e < e1; e += 256)      // .set semantics: plain store
    S[lsrc[e]][ltgt[e]] = eb[e * Hc + h];

#ifdef HAVE_TDM
  if (tid < 32) __builtin_amdgcn_s_wait_tensorcnt(0);   // drain TENSORcnt
#endif
  __syncthreads();                                // K/V^T now visible to all

  const int s0 = wave * 16;
  const int ttiles = (cnt + 15) >> 4;
  if (s0 < cnt) {
    v16h a0 = ldfrag_row_f16(Qb + (size_t)s0 * HDc, HDc, lane);
    v16h a1 = ldfrag_row_f16(Qb + (size_t)s0 * HDc + 32, HDc, lane);
    for (int tt = 0; tt < ttiles; ++tt) {
      v8f acc = v8f_zero();
      v16h b0 = ldfrag_row_f16(Ksh + (size_t)tt * 16 * HDc, HDc, lane);
      acc = WMMA16(a0, b0, acc);
      v16h b1 = ldfrag_row_f16(Ksh + (size_t)tt * 16 * HDc + 32, HDc, lane);
      acc = WMMA16(a1, b1, acc);
      const int col = tt * 16 + (lane & 15), hi = lane >> 4;
#pragma unroll
      for (int i = 0; i < 8; ++i)
        S[s0 + i + 8 * hi][col] += acc[i] * 0.125f;   // /= sqrt(hd)=8
    }
  }
  __syncthreads();

  // softmax: row r handled by thread pair (2r, 2r+1), each does 64 cols
  {
    const int r = tid >> 1, c0 = (tid & 1) * 64;
    if (r < cnt) {
      float m = -3.0e38f;
      for (int c = c0; c < c0 + 64; ++c)
        if (c < cnt) m = fmaxf(m, S[r][c]);
      m = fmaxf(m, __shfl_xor(m, 1, 32));
      float sum = 0.f;
      for (int c = c0; c < c0 + 64; ++c) {
        float p = (c < cnt) ? __expf(S[r][c] - m) : 0.f;
        S[r][c] = p;
        sum += p;
      }
      sum += __shfl_xor(sum, 1, 32);
      float inv = (sum > 0.f) ? 1.f / sum : 0.f;
      for (int c = c0; c < c0 + 64; ++c) P16[r][c] = (half_t)(S[r][c] * inv);
    } else {
      for (int c = c0; c < c0 + 64; ++c) P16[r][c] = (half_t)0.f;
    }
  }
  __syncthreads();

  if (s0 < cnt) {
    const int ksteps = (cnt + 31) >> 5;
    v8f acc[4] = {v8f_zero(), v8f_zero(), v8f_zero(), v8f_zero()};
    for (int ks = 0; ks < ksteps; ++ks) {
      v16h a = ldfrag_row_f16(&P16[s0][ks * 32], MAXCc, lane);
#pragma unroll
      for (int j = 0; j < 4; ++j) {
        // B[k=t][n=d] = V^T[d][t]: row-major fragment on Vtsh (ld = MAXC)
        v16h b = ldfrag_row_f16(Vtsh + (size_t)(j * 16) * MAXCc + ks * 32, MAXCc, lane);
        acc[j] = WMMA16(a, b, acc[j]);
      }
    }
    const int col16 = lane & 15, hi = lane >> 4;
#pragma unroll
    for (int j = 0; j < 4; ++j)
#pragma unroll
      for (int i = 0; i < 8; ++i) {
        const int s = s0 + i + 8 * hi;
        opad[((size_t)g * MAXCc + s) * Dc + h * HDc + j * 16 + col16] =
            (half_t)acc[j][i];
      }
  }
}

// ---------------------------------------------------------------------------
// out[n] = gather(opad)[n] @ w_out^T + b_out + features[n]   (WMMA, f32 out)
// ---------------------------------------------------------------------------
__global__ __launch_bounds__(256) void out_gemm_kernel(
    const half_t* __restrict__ opad, const half_t* __restrict__ wout,
    const float* __restrict__ bout, const float* __restrict__ features,
    const int* __restrict__ batch_map, const int* __restrict__ pos_in_group,
    float* __restrict__ out) {
  const int nstrips = Dc / 64;                   // 8
  const int strip = blockIdx.x % nstrips;
  const int mblk  = blockIdx.x / nstrips;
  const int wave = threadIdx.x >> 5, lane = threadIdx.x & 31;
  const int m0 = mblk * 128 + wave * 16;
  const int n0 = strip * 64;

  const int myrow = m0 + (lane & 15);            // ragged row gathered by lane
  const size_t abase =
      ((size_t)batch_map[myrow] * MAXCc + pos_in_group[myrow]) * Dc;

  v8f acc[4] = {v8f_zero(), v8f_zero(), v8f_zero(), v8f_zero()};
  for (int k = 0; k < Dc; k += 32) {
    __builtin_prefetch(opad + abase + k + 64, 0, 1);
    v16h a;
    {
      const half_t* p = opad + abase + k + ((lane >> 4) << 3);
#pragma unroll
      for (int i = 0; i < 8; ++i) { a[i] = p[i]; a[i + 8] = p[i + 16]; }
    }
#pragma unroll
    for (int j = 0; j < 4; ++j) {
      v16h b = ldfrag_row_f16(wout + (size_t)(n0 + j * 16) * Dc + k, Dc, lane);
      acc[j] = WMMA16(a, b, acc[j]);
    }
  }
  const int col16 = lane & 15, hi = lane >> 4;
#pragma unroll
  for (int j = 0; j < 4; ++j)
#pragma unroll
    for (int i = 0; i < 8; ++i) {
      const int m = m0 + i + 8 * hi;
      const int jj = n0 + j * 16 + col16;
      out[(size_t)m * Dc + jj] =
          acc[j][i] + bout[jj] + features[(size_t)m * Dc + jj];
    }
}

// ---------------------------------------------------------------------------
extern "C" void kernel_launch(void* const* d_in, const int* in_sizes, int n_in,
                              void* d_out, int out_size, void* d_ws, size_t ws_size,
                              hipStream_t stream) {
  const float* features    = (const float*)d_in[0];
  const float* edge_radial = (const float*)d_in[1];
  const int*   batch_map   = (const int*)d_in[2];
  const int*   pos_in_grp  = (const int*)d_in[3];
  // d_in[4] = edge_batch (unused; edges are contiguous per group)
  const int*   local_src   = (const int*)d_in[5];
  const int*   local_tgt   = (const int*)d_in[6];
  const unsigned char* mask = (const unsigned char*)d_in[7];
  const float* ln1_g = (const float*)d_in[8];
  const float* ln1_b = (const float*)d_in[9];
  const float* w_kqv = (const float*)d_in[10];
  const float* lnb_g = (const float*)d_in[11];
  const float* lnb_b = (const float*)d_in[12];
  const float* w_bias = (const float*)d_in[13];
  const float* b_bias = (const float*)d_in[14];
  const float* w_out  = (const float*)d_in[15];
  const float* b_out  = (const float*)d_in[16];
  float* out = (float*)d_out;

  const long N = in_sizes[0] / Dc;               // 21504
  const long E = in_sizes[4];                    // 688128

  // --- workspace carve-out (256B aligned) ---
  char* ws = (char*)d_ws;
  size_t off = 0;
  auto alloc = [&](size_t bytes) {
    void* p = ws + off;
    off = (off + bytes + 255) & ~(size_t)255;
    return p;
  };
  half_t* wkqv_h = (half_t*)alloc((size_t)KQVc * Dc * sizeof(half_t));
  half_t* wout_h = (half_t*)alloc((size_t)Dc * Dc * sizeof(half_t));
  half_t* xln    = (half_t*)alloc((size_t)N * Dc * sizeof(half_t));
  half_t* kpad   = (half_t*)alloc((size_t)Gc * Hc * MAXCc * HDc * sizeof(half_t));
  half_t* qpad   = (half_t*)alloc((size_t)Gc * Hc * MAXCc * HDc * sizeof(half_t));
  half_t* vpadT  = (half_t*)alloc((size_t)Gc * Hc * HDc * MAXCc * sizeof(half_t));
  float*  ebuf   = (float*)alloc((size_t)E * Hc * sizeof(float));
  half_t* opad   = (half_t*)alloc((size_t)Gc * MAXCc * Dc * sizeof(half_t));
  int* counts  = (int*)alloc(Gc * sizeof(int));
  int* offsets = (int*)alloc(Gc * sizeof(int));

  // 1. group sizes + offsets from mask
  counts_offsets_kernel<<<1, 256, 0, stream>>>(mask, counts, offsets);
  // 2. weight conversion to f16
  {
    long n1 = (long)KQVc * Dc, n2 = (long)Dc * Dc;
    f32_to_f16_kernel<<<(int)((n1 + 255) / 256), 256, 0, stream>>>(w_kqv, wkqv_h, n1);
    f32_to_f16_kernel<<<(int)((n2 + 255) / 256), 256, 0, stream>>>(w_out, wout_h, n2);
  }
  // 3. LayerNorm(features) -> f16
  ln_features_kernel<<<(int)N, 256, 0, stream>>>(features, ln1_g, ln1_b, xln);
  // 4. kqv GEMM + scatter to padded layout (V transposed)
  {
    int blocks = (int)(N / 128) * (KQVc / 64);   // 168 * 24
    kqv_gemm_scatter_kernel<<<blocks, 256, 0, stream>>>(
        xln, wkqv_h, batch_map, pos_in_grp, kpad, qpad, vpadT);
  }
  // 5. edge radial bias -> eb[E,H]
  edge_bias_kernel<<<(int)((E + 7) / 8), 256, 0, stream>>>(
      edge_radial, lnb_g, lnb_b, w_bias, b_bias, ebuf, E);
  // 6. per-(group,head) attention (TDM-staged K / V^T)
  attention_kernel<<<Gc * Hc, 256, 0, stream>>>(
      qpad, kpad, vpadT, ebuf, local_src, local_tgt, counts, offsets, opad);
  // 7. gathered out-projection + residual
  {
    int blocks = (int)(N / 128) * (Dc / 64);     // 168 * 8
    out_gemm_kernel<<<blocks, 256, 0, stream>>>(
        opad, wout_h, b_out, features, batch_map, pos_in_grp, out);
  }
}
